// Decoder_48593259987147
// MI455X (gfx1250) — matrix operand order repaired and checked by reading, not compiled, not run on hardware
//
#include <hip/hip_runtime.h>
#include <hip/hip_bf16.h>
#include <math.h>

typedef __bf16 bf16;
typedef __attribute__((ext_vector_type(16))) __bf16 v16bf;
typedef __attribute__((ext_vector_type(8)))  float  v8f;

#define kB  32
#define kT  256
#define kF  512
#define kH  1600
#define kA  60
#define G4H 6400          // 4*kH
#define KC  2112          // kF + kH (combined [W_x; W_h] K extent)

// ---------------------------------------------------------------------------
// WMMA fragment loaders (CDNA5 wave32 layouts, ISA 7.12.2)
// A 16x32 bf16, row-major source, stride lda:
//   lane<16 : row=lane,    K = {0..7} in v[0..7], {16..23} in v[8..15]
//   lane>=16: row=lane-16, K = {8..15},           {24..31}
__device__ __forceinline__ v16bf load_a16x32(const bf16* base, int lda, int lane) {
    const bf16* rp = base + (size_t)(lane & 15) * lda + ((lane >> 4) << 3);
    v16bf a;
#pragma unroll
    for (int e = 0; e < 8; ++e) { a[e] = rp[e]; a[e + 8] = rp[16 + e]; }
    return a;
}
// B 32x16 bf16 from a TRANSPOSED source BT[n][k] (rows = N, contiguous K), stride ldb:
//   lane<16 : N=lane,    K = {0..7},{16..23} ; lane>=16: N=lane-16, K = {8..15},{24..31}
__device__ __forceinline__ v16bf load_bT32x16(const bf16* baseT, int ldb, int lane) {
    const bf16* cp = baseT + (size_t)(lane & 15) * ldb + ((lane >> 4) << 3);
    v16bf b;
#pragma unroll
    for (int e = 0; e < 8; ++e) { b[e] = cp[e]; b[e + 8] = cp[16 + e]; }
    return b;
}

// ---------------------------------------------------------------------------
// Conversions / packing
__global__ void k_f2bf(const float* __restrict__ in, bf16* __restrict__ out, int n) {
    int i = blockIdx.x * blockDim.x + threadIdx.x;
    if (i < n) out[i] = (bf16)in[i];
}
// W_cT[n][k] = k<F ? W_x[k][n] : W_h[k-F][n]   (bf16, column-contiguous for B frags)
__global__ void k_make_wcT(const float* __restrict__ Wx, const float* __restrict__ Wh,
                           bf16* __restrict__ WcT) {
    int i = blockIdx.x * blockDim.x + threadIdx.x;
    if (i >= G4H * KC) return;
    int n = i / KC, k = i - n * KC;
    float v = (k < kF) ? Wx[(size_t)k * G4H + n] : Wh[(size_t)(k - kF) * G4H + n];
    WcT[i] = (bf16)v;
}
// HvT[b][d][s] = H_value[b][s][d]
__global__ void k_make_hvT(const float* __restrict__ Hvf, bf16* __restrict__ HvT) {
    int i = blockIdx.x * blockDim.x + threadIdx.x;
    if (i >= kB * kH * kT) return;
    int b = i / (kH * kT), r = i - b * kH * kT;
    int d = r / kT, s = r - d * kT;
    HvT[i] = (bf16)Hvf[((size_t)b * kT + s) * kH + d];
}
// initial state: h=s0, c=c0, S_key[:,0,:]=s0
__global__ void k_init(const float* __restrict__ s0, const float* __restrict__ c0,
                       float* __restrict__ h, float* __restrict__ c,
                       bf16* __restrict__ h_bf, bf16* __restrict__ Skey) {
    int i = blockIdx.x * blockDim.x + threadIdx.x;
    if (i >= kB * kH) return;
    int b = i / kH, j = i - b * kH;
    float hv = s0[i];
    h[i] = hv; c[i] = c0[i];
    h_bf[i] = (bf16)hv;
    Skey[((size_t)b * kT) * kH + j] = (bf16)hv;
}

// ---------------------------------------------------------------------------
// Step-independent pre-pass: xzT[t][n][m] = sum_{k<512} y[m][t][k] * W_x[k][n].
// Roles swapped (A = W_cT n-rows, B = y rows as columns) so the WMMA tile comes
// out already transposed to [n][m] -- which makes the recurrent kernel's
// accumulator init a contiguous 16-byte load per lane.
__global__ void __launch_bounds__(256) k_xz_gemm(
        const bf16* __restrict__ y_bf, const bf16* __restrict__ WcT,
        bf16* __restrict__ xzT) {
    int lane = threadIdx.x & 31;
    int gw = blockIdx.x * (blockDim.x >> 5) + (threadIdx.x >> 5);
    if (gw >= kT * 2 * (G4H / 16)) return;
    int mt = gw & 1;                    // m-tile (batch 0..15 / 16..31)
    int nt = (gw >> 1) % (G4H / 16);    // n-tile 0..399
    int t  = gw / (2 * (G4H / 16));     // time step
    int m0 = mt * 16, n0 = nt * 16;
    v8f acc = {};
    const bf16* arow = WcT + (size_t)n0 * KC;                    // A: 16 n-rows
    const bf16* bcol = y_bf + ((size_t)m0 * kT + t) * kF;        // B: cols = m
    for (int kt2 = 0; kt2 < kF / 32; ++kt2) {
        v16bf a  = load_a16x32(arow + kt2 * 32, KC, lane);
        v16bf bb = load_bT32x16(bcol + kt2 * 32, kT * kF, lane);
        acc = __builtin_amdgcn_wmma_f32_16x16x32_bf16(false, a, false, bb,
                                                      (short)0, acc, false, false);
    }
    int col = lane & 15, r0 = (lane >> 4) * 8;                   // col = m offset
#pragma unroll
    for (int v = 0; v < 8; ++v)
        xzT[((size_t)t * G4H + n0 + r0 + v) * kB + m0 + col] = (bf16)acc[v];
}

// ---------------------------------------------------------------------------
// Fused recurrent step. Block jt owns hidden columns [jt*16, jt*16+16) across
// ALL FOUR gates: 8 waves = {m-tile 0,1} x {i,f,g,o}, one 16x16 WMMA tile each.
// Accumulator starts from precomputed xzT[t]; K-loop covers ONLY h @ W_h
// (branch-free single address stream -> pipelinable). z stays in LDS.
__global__ void __launch_bounds__(256) k_lstm_step(
        const bf16* __restrict__ xzT, const bf16* __restrict__ h_bf_in,
        const bf16* __restrict__ WcT, const float* __restrict__ bl,
        float* __restrict__ h, float* __restrict__ c,
        bf16* __restrict__ h_bf_out, float* __restrict__ S,
        bf16* __restrict__ Skey, int t) {
    __shared__ float zbuf[4][kB][16];          // 8 KB: [gate][row(batch)][col(j)]
    int lane = threadIdx.x & 31;
    int w    = threadIdx.x >> 5;               // 0..7
    int mt   = w & 1;                          // m-tile
    int g    = w >> 1;                         // gate 0..3 (i,f,g,o)
    int jt   = blockIdx.x;                     // 0..99 hidden-column tile
    int m0   = mt * 16;
    int n0   = g * kH + jt * 16;               // column in the 4H gate space
    int col  = lane & 15, r0 = (lane >> 4) * 8;

    // init accumulator from the precomputed y_t @ W_x tile (contiguous per lane)
    v8f acc;
    {
        const bf16* xp = xzT + ((size_t)t * G4H + n0 + col) * kB + m0 + r0;
#pragma unroll
        for (int v = 0; v < 8; ++v) acc[v] = (float)xp[v];
    }

    const bf16* wrow = WcT + (size_t)n0 * KC + kF;      // recurrent K range only
    const bf16* ah   = h_bf_in + (size_t)m0 * kH;
    const bf16* pfp  = wrow + (size_t)col * KC;         // this lane's weight row
#pragma unroll 2
    for (int kt2 = 0; kt2 < kH / 32; ++kt2) {
        int k0 = kt2 * 32;
        __builtin_prefetch(pfp + k0 + 128, 0, 3);       // near-scope, 256B ahead
        v16bf a  = load_a16x32(ah + k0, kH, lane);
        v16bf bb = load_bT32x16(wrow + k0, KC, lane);
        acc = __builtin_amdgcn_wmma_f32_16x16x32_bf16(false, a, false, bb,
                                                      (short)0, acc, false, false);
    }
#pragma unroll
    for (int v = 0; v < 8; ++v) zbuf[g][m0 + r0 + v][col] = acc[v];
    __syncthreads();

    // Gate nonlinearity: 512 elements (32 batch x 16 cols), 2 per thread.
#pragma unroll
    for (int e = 0; e < 2; ++e) {
        int idx = threadIdx.x + e * 256;       // 0..511
        int m = idx >> 4, cc = idx & 15;       // batch row, column within tile
        int j = jt * 16 + cc;                  // hidden index
        int gi = m * kH + j;                   // state index
        float iv = zbuf[0][m][cc] + bl[j];
        float fv = zbuf[1][m][cc] + bl[kH + j];
        float gv = zbuf[2][m][cc] + bl[2 * kH + j];
        float ov = zbuf[3][m][cc] + bl[3 * kH + j];
        float si = 1.f / (1.f + expf(-iv));
        float sf = 1.f / (1.f + expf(-fv));
        float so = 1.f / (1.f + expf(-ov));
        float cn = sf * c[gi] + si * tanhf(gv);
        float hn = so * tanhf(cn);
        c[gi] = cn; h[gi] = hn;
        h_bf_out[gi] = (bf16)hn;
        S[((size_t)m * kT + t) * kH + j] = hn;
        if (t + 1 < kT) Skey[((size_t)m * kT + t + 1) * kH + j] = (bf16)hn;
    }
}

// ---------------------------------------------------------------------------
// Attention: one block per (b, 16-row tile). scores+softmax+P in LDS.
__global__ void __launch_bounds__(256) k_attention(
        const bf16* __restrict__ Skey, const bf16* __restrict__ Hv,
        const bf16* __restrict__ HvT, float* __restrict__ Att) {
    __shared__ float sc[16 * kT];
    __shared__ bf16  pl[16 * kT];
    int b = blockIdx.x >> 4, tt = blockIdx.x & 15, t0 = tt * 16;
    int lane = threadIdx.x & 31, wid = threadIdx.x >> 5;

    // scores[16 x 256] = S_key_tile @ H_value^T
    const bf16* Abase = Skey + ((size_t)b * kT + t0) * kH;
    for (int nt = wid; nt < kT / 16; nt += 8) {
        int n0 = nt * 16;
        const bf16* Bbase = Hv + ((size_t)b * kT + n0) * kH;  // row n = s, contiguous d
        v8f acc = {};
        for (int kt2 = 0; kt2 < kH / 32; ++kt2) {
            v16bf a  = load_a16x32(Abase + kt2 * 32, kH, lane);
            v16bf bb = load_bT32x16(Bbase + kt2 * 32, kH, lane);
            acc = __builtin_amdgcn_wmma_f32_16x16x32_bf16(false, a, false, bb,
                                                          (short)0, acc, false, false);
        }
        int col = lane & 15, r0 = (lane >> 4) * 8;
#pragma unroll
        for (int v = 0; v < 8; ++v) sc[(r0 + v) * kT + n0 + col] = acc[v];
    }
    __syncthreads();

    // row softmax -> P (bf16) in LDS
    if (threadIdx.x < 16) {
        float* row = sc + threadIdx.x * kT;
        float m = row[0];
        for (int s = 1; s < kT; ++s) m = fmaxf(m, row[s]);
        float sum = 0.f;
        for (int s = 0; s < kT; ++s) { float e = expf(row[s] - m); row[s] = e; sum += e; }
        float inv = 1.f / sum;
        bf16* prow = pl + threadIdx.x * kT;
        for (int s = 0; s < kT; ++s) prow[s] = (bf16)(row[s] * inv);
    }
    __syncthreads();

    // Att_tile[16 x 1600] = P @ H_value  (B from HvT, contiguous s per d-row)
    for (int nt = wid; nt < kH / 16; nt += 8) {
        int n0 = nt * 16;
        const bf16* BT = HvT + ((size_t)b * kH + n0) * kT;
        v8f acc = {};
        for (int kt2 = 0; kt2 < kT / 32; ++kt2) {
            v16bf a  = load_a16x32(pl + kt2 * 32, kT, lane);
            v16bf bb = load_bT32x16(BT + kt2 * 32, kT, lane);
            acc = __builtin_amdgcn_wmma_f32_16x16x32_bf16(false, a, false, bb,
                                                          (short)0, acc, false, false);
        }
        int col = lane & 15, r0 = (lane >> 4) * 8;
#pragma unroll
        for (int v = 0; v < 8; ++v)
            Att[((size_t)b * kT + t0 + r0 + v) * kH + n0 + col] = acc[v];
    }
}

// ---------------------------------------------------------------------------
// Dense -> softmax(60) -> dihedrals.  fp32 for the precision-sensitive tail.
__global__ void __launch_bounds__(64) k_dense(
        const float* __restrict__ S, const float* __restrict__ Att,
        const float* __restrict__ Wd, const float* __restrict__ bd,
        const float* __restrict__ alpha, float* __restrict__ dih) {
    __shared__ float srow[2 * kH];
    __shared__ float xv[kA];
    int row = blockIdx.x;                       // b*T + t
    const float* Srow = S   + (size_t)row * kH;
    const float* Arow = Att + (size_t)row * kH;
    for (int k = threadIdx.x; k < kH; k += blockDim.x) {
        srow[k] = Srow[k]; srow[kH + k] = Arow[k];
    }
    __syncthreads();
    if (threadIdx.x < kA) {
        int a = threadIdx.x;
        float acc = bd[a];
        for (int k = 0; k < 2 * kH; ++k) acc += srow[k] * Wd[(size_t)k * kA + a];
        xv[a] = acc;
    }
    __syncthreads();
    if (threadIdx.x == 0) {
        float m = xv[0];
        for (int a = 1; a < kA; ++a) m = fmaxf(m, xv[a]);
        float w[kA]; float sum = 0.f;
        for (int a = 0; a < kA; ++a) { w[a] = expf(xv[a] - m); sum += w[a]; }
        float inv = 1.f / sum;
        for (int d = 0; d < 3; ++d) {
            float sy = 0.f, sx = 0.f;
            for (int a = 0; a < kA; ++a) {
                float ang = alpha[a * 3 + d];
                sy += w[a] * inv * sinf(ang);
                sx += w[a] * inv * cosf(ang);
            }
            dih[(size_t)row * 3 + d] = atan2f(sy, sx);
        }
    }
}

// ---------------------------------------------------------------------------
// Sequential NeRF extension: one wave, lane j = batch column (reshape-faithful).
__global__ void k_geometry(const float* __restrict__ dih, float* __restrict__ out) {
    int j = threadIdx.x;
    if (j >= kB) return;
    const float PI = 3.14159265358979f;
    const float BL[3] = {145.801f, 152.326f, 132.868f};
    const float BA[3] = {2.124f, 1.941f, 2.028f};
    float rc[3], rs[3];
#pragma unroll
    for (int d = 0; d < 3; ++d) {
        rc[d] = BL[d] * cosf(PI - BA[d]);
        rs[d] = BL[d] * sinf(PI - BA[d]);
    }
    float ax = -sqrtf(0.5f), ay = sqrtf(1.5f), az = 0.f;
    float bx = -sqrtf(2.0f), by = 0.f,         bz = 0.f;
    float cx = 0.f,          cy = 0.f,         cz = 0.f;
    for (int l = 0; l < kT * 3; ++l) {
        int i = l / 3, d = l - i * 3;
        float phi = dih[((size_t)i * kB + j) * 3 + d];
        float px = rc[d], py = cosf(phi) * rs[d], pz = sinf(phi) * rs[d];
        float ux = cx - bx, uy = cy - by, uz = cz - bz;
        float ui = 1.f / sqrtf(fmaxf(ux * ux + uy * uy + uz * uz, 1e-12f));
        ux *= ui; uy *= ui; uz *= ui;
        float vx = bx - ax, vy = by - ay, vz = bz - az;
        float nx = vy * uz - vz * uy, ny = vz * ux - vx * uz, nz = vx * uy - vy * ux;
        float ni = 1.f / sqrtf(fmaxf(nx * nx + ny * ny + nz * nz, 1e-12f));
        nx *= ni; ny *= ni; nz *= ni;
        float mx = ny * uz - nz * uy, my = nz * ux - nx * uz, mz = nx * uy - ny * ux;
        float ox = ux * px + mx * py + nx * pz + cx;
        float oy = uy * px + my * py + ny * pz + cy;
        float oz = uz * px + mz * py + nz * pz + cz;
        ax = bx; ay = by; az = bz;
        bx = cx; by = cy; bz = cz;
        cx = ox; cy = oy; cz = oz;
        size_t o = ((size_t)l * kB + j) * 3;
        out[o] = ox; out[o + 1] = oy; out[o + 2] = oz;
    }
}

__global__ void k_copy_hc(const float* __restrict__ h, const float* __restrict__ c,
                          float* __restrict__ out) {
    int i = blockIdx.x * blockDim.x + threadIdx.x;
    const int n = kB * kH;
    if (i < n) out[i] = h[i];
    else if (i < 2 * n) out[i] = c[i - n];
}

// ---------------------------------------------------------------------------
extern "C" void kernel_launch(void* const* d_in, const int* in_sizes, int n_in,
                              void* d_out, int out_size, void* d_ws, size_t ws_size,
                              hipStream_t stream) {
    const float* y   = (const float*)d_in[0];
    const float* s0  = (const float*)d_in[1];
    const float* c0i = (const float*)d_in[2];
    const float* Hvf = (const float*)d_in[3];
    const float* Wx  = (const float*)d_in[4];
    const float* Wh  = (const float*)d_in[5];
    const float* bl  = (const float*)d_in[6];
    const float* Wd  = (const float*)d_in[7];
    const float* bd  = (const float*)d_in[8];
    const float* alp = (const float*)d_in[9];
    float* out = (float*)d_out;

    char* p = (char*)d_ws;
    auto alloc = [&](size_t bytes) -> void* {
        void* r = (void*)p;
        p += (bytes + 255) & ~(size_t)255;
        return r;
    };
    bf16*  WcT  = (bf16*) alloc((size_t)G4H * KC * 2);           //  27.0 MB
    bf16*  y_bf = (bf16*) alloc((size_t)kB * kT * kF * 2);       //   8.4 MB
    bf16*  Hv   = (bf16*) alloc((size_t)kB * kT * kH * 2);       //  26.2 MB
    bf16*  HvT  = (bf16*) alloc((size_t)kB * kH * kT * 2);       //  26.2 MB
    bf16*  Skey = (bf16*) alloc((size_t)kB * kT * kH * 2);       //  26.2 MB
    float* S    = (float*)alloc((size_t)kB * kT * kH * 4);       //  52.4 MB
    float* Att  = (float*)alloc((size_t)kB * kT * kH * 4);       //  52.4 MB
    bf16*  xzT  = (bf16*) alloc((size_t)kT * G4H * kB * 2);      // 104.9 MB
    float* hws  = (float*)alloc((size_t)kB * kH * 4);
    float* cws  = (float*)alloc((size_t)kB * kH * 4);
    bf16*  h_bf = (bf16*) alloc((size_t)kB * kH * 2);
    float* dih  = (float*)alloc((size_t)kB * kT * 3 * 4);

    // pack / transpose / convert
    k_f2bf<<<(kB * kT * kF + 255) / 256, 256, 0, stream>>>(y, y_bf, kB * kT * kF);
    k_f2bf<<<(kB * kT * kH + 255) / 256, 256, 0, stream>>>(Hvf, Hv, kB * kT * kH);
    k_make_hvT<<<(kB * kH * kT + 255) / 256, 256, 0, stream>>>(Hvf, HvT);
    k_make_wcT<<<(G4H * KC + 255) / 256, 256, 0, stream>>>(Wx, Wh, WcT);
    k_init<<<(kB * kH + 255) / 256, 256, 0, stream>>>(s0, c0i, hws, cws, h_bf, Skey);

    // Step-independent y @ W_x for ALL steps (fully parallel, 204800 tiles)
    k_xz_gemm<<<kT * 2 * (G4H / 16) / 8, 256, 0, stream>>>(y_bf, WcT, xzT);

    // LSTM recurrence: one fused kernel/step; K-loop = recurrent part only.
    // W_h columns (20.5 MB bf16) stay L2-resident across all 256 steps.
    for (int t = 0; t < kT; ++t) {
        k_lstm_step<<<kH / 16, 256, 0, stream>>>(xzT, h_bf, WcT, bl,
                                                 hws, cws, h_bf, S, Skey, t);
    }

    k_attention<<<kB * (kT / 16), 256, 0, stream>>>(Skey, Hv, HvT, Att);
    k_dense<<<kB * kT, 64, 0, stream>>>(S, Att, Wd, bd, alp, dih);
    k_geometry<<<1, 32, 0, stream>>>(dih, out);
    k_copy_hc<<<(2 * kB * kH + 255) / 256, 256, 0, stream>>>(
        hws, cws, out + (size_t)kT * 3 * kB * 3);
}
